// CodebookLayer_58394375357111
// MI455X (gfx1250) — matrix operand order repaired and checked by reading, not compiled, not run on hardware
//
#include <hip/hip_runtime.h>

// ---------------------------------------------------------------------------
// Nearest-codebook snap (top-1 L2) for MI455X / gfx1250.
//   x: [M=8192, D=1024] f32, codebook: [C=8192, D=1024] f32
//   out[m] = codebook[argmax_c (2*x.cb - ||cb_c||^2)]
// Core: bf16 WMMA GEMM (v_wmma_f32_16x16x32_bf16), pre-converted bf16 operands,
// async global->LDS double-buffered staging, fused argmax epilogue.
// ---------------------------------------------------------------------------

typedef __attribute__((ext_vector_type(16))) __bf16 v16bf;
typedef __attribute__((ext_vector_type(8)))  float  v8f;

#define TILE   128   // M and N tile
#define TK     32    // K step (bf16 WMMA depth)
#define LDSTR  40    // padded LDS row stride in bf16 elements (80 B)
#define BUFB   (TILE * LDSTR * 2)   // bytes per LDS tile buffer (10240)

// round-to-nearest-even f32 -> bf16 (inputs are finite random normals)
static __device__ __forceinline__ unsigned f2bf(float f) {
    unsigned u = __float_as_uint(f);
    u += 0x7FFFu + ((u >> 16) & 1u);
    return u >> 16;
}

// monotone float -> uint key (order-preserving over all finite floats)
static __device__ __forceinline__ unsigned fkey(float f) {
    unsigned u = __float_as_uint(f);
    return (u & 0x80000000u) ? ~u : (u | 0x80000000u);
}

static __device__ __forceinline__ uint4 pack8(float4 a, float4 b) {
    uint4 r;
    r.x = f2bf(a.x) | (f2bf(a.y) << 16);
    r.y = f2bf(a.z) | (f2bf(a.w) << 16);
    r.z = f2bf(b.x) | (f2bf(b.y) << 16);
    r.w = f2bf(b.z) | (f2bf(b.w) << 16);
    return r;
}

// generic LDS pointer -> 32-bit LDS byte address (low 32 bits of LDS aperture)
static __device__ __forceinline__ unsigned lds_addr(const void* p) {
    return (unsigned)(unsigned long long)p;
}

// async 16-byte global->LDS copy (per lane), GVS addressing: s[base]+voff
static __device__ __forceinline__ void async_ld16(unsigned lds, unsigned goff,
                                                  const void* base) {
    asm volatile("global_load_async_to_lds_b128 %0, %1, %2"
                 :: "v"(lds), "v"(goff), "s"(base) : "memory");
}
#define WAIT_ASYNC(n) asm volatile("s_wait_asynccnt %0" :: "i"(n) : "memory")

// ---------------------------------------------------------------------------
// Kernel 0: f32 -> bf16 bulk conversion (one uint4 = 8 elements per thread)
// ---------------------------------------------------------------------------
__global__ void cvt_bf16_kernel(const float* __restrict__ src,
                                uint4* __restrict__ dst) {
    const size_t gid = (size_t)blockIdx.x * blockDim.x + threadIdx.x;
    const float4* s = (const float4*)src;
    float4 a = s[gid * 2], b = s[gid * 2 + 1];
    dst[gid] = pack8(a, b);
}

// ---------------------------------------------------------------------------
// Kernel 1: c2[c] = ||codebook[c]||^2 ; also init best[] (one block per row)
// ---------------------------------------------------------------------------
__global__ void c2_init_kernel(const float* __restrict__ cb,
                               float* __restrict__ c2,
                               unsigned long long* __restrict__ best,
                               int D, int Mrows) {
    const int row = blockIdx.x;
    const float4* p = (const float4*)(cb + (size_t)row * D);
    float4 v = p[threadIdx.x];                 // 256 thr * 4 = 1024 = D
    float s = v.x * v.x + v.y * v.y + v.z * v.z + v.w * v.w;
    #pragma unroll
    for (int m = 16; m; m >>= 1) s += __shfl_xor(s, m, 32);
    __shared__ float wsum[8];
    if ((threadIdx.x & 31) == 0) wsum[threadIdx.x >> 5] = s;
    __syncthreads();
    if (threadIdx.x == 0) {
        float t = 0.f;
        #pragma unroll
        for (int i = 0; i < 8; ++i) t += wsum[i];
        c2[row] = t;
        if (row < Mrows) best[row] = 0ull;     // any finite score beats key 0
    }
}

// ---------------------------------------------------------------------------
// Shared epilogue: fused argmax of score = 2*acc - c2[col]
// ---------------------------------------------------------------------------
static __device__ __forceinline__ void argmax_epilogue(
    v8f acc[4][2], const float* __restrict__ c2,
    unsigned long long* __restrict__ best, unsigned long long* sBest,
    int m0, int n0, int wm, int wn, int hl, int l15, int t) {
    int cols[2];
    float c2v[2];
    #pragma unroll
    for (int j = 0; j < 2; ++j) {
        cols[j] = n0 + wn * 32 + j * 16 + l15;
        c2v[j]  = c2[cols[j]];
    }
    #pragma unroll
    for (int i = 0; i < 4; ++i) {
        #pragma unroll
        for (int r = 0; r < 8; ++r) {
            // pack (monotone key << 32) | ~col : max => best score, lowest idx
            unsigned long long pk = 0ull;
            #pragma unroll
            for (int j = 0; j < 2; ++j) {
                float score = 2.0f * acc[i][j][r] - c2v[j];
                unsigned long long p =
                    ((unsigned long long)fkey(score) << 32) |
                    (unsigned)(~(unsigned)cols[j]);
                if (p > pk) pk = p;
            }
            // butterfly max across the 16 lanes sharing this row
            #pragma unroll
            for (int m = 8; m; m >>= 1) {
                unsigned hi = (unsigned)(pk >> 32), lo = (unsigned)pk;
                unsigned ohi = __shfl_xor(hi, m, 32);
                unsigned olo = __shfl_xor(lo, m, 32);
                unsigned long long o = ((unsigned long long)ohi << 32) | olo;
                if (o > pk) pk = o;
            }
            if (l15 == 0)
                atomicMax(&sBest[wm * 64 + i * 16 + hl * 8 + r], pk);
        }
    }
    __syncthreads();
    if (t < TILE) atomicMax(&best[m0 + t], sBest[t]);
}

// ---------------------------------------------------------------------------
// Kernel 2 (primary): bf16-in GEMM + argmax, async double-buffered staging
// grid = (C/128, M/128), block = 256 (8 waves); wave: 4x2 16x16 WMMA tiles
// ---------------------------------------------------------------------------
__global__ void __launch_bounds__(256)
snap_gemm_argmax_bf16(const unsigned short* __restrict__ Xb,
                      const unsigned short* __restrict__ CBb,
                      const float* __restrict__ c2,
                      unsigned long long* __restrict__ best, int D) {
    __shared__ unsigned short sA[2][TILE * LDSTR];
    __shared__ unsigned short sB[2][TILE * LDSTR];
    __shared__ unsigned long long sBest[TILE];

    const int t    = threadIdx.x;
    const int lane = t & 31;
    const int w    = t >> 5;
    const int wm   = w >> 2;        // 0..1
    const int wn   = w & 3;         // 0..3
    const int hl   = lane >> 4;
    const int l15  = lane & 15;

    const int m0 = blockIdx.y * TILE;
    const int n0 = blockIdx.x * TILE;

    if (t < TILE) sBest[t] = 0ull;

    v8f acc[4][2] = {};

    // staging: each thread async-copies 2x16B of A and of B per K step
    const int srow = t >> 1;                       // 0..127
    const int scol = (t & 1) * 16;                 // 0 or 16 (elements)
    const unsigned gaBase = ((unsigned)(m0 + srow) * (unsigned)D + scol) * 2u;
    const unsigned gbBase = ((unsigned)(n0 + srow) * (unsigned)D + scol) * 2u;
    const unsigned laBase = lds_addr(&sA[0][srow * LDSTR + scol]);
    const unsigned lbBase = lds_addr(&sB[0][srow * LDSTR + scol]);

    const int NK = D / TK;

    // prologue: stage K-step 0 into buffer 0
    {
        unsigned ga = gaBase, gb = gbBase;
        async_ld16(laBase,      ga,      Xb);
        async_ld16(laBase + 16, ga + 16, Xb);
        async_ld16(lbBase,      gb,      CBb);
        async_ld16(lbBase + 16, gb + 16, CBb);
    }

    for (int it = 0; it < NK; ++it) {
        const int buf = it & 1;
        if (it + 1 < NK) {
            // stage next K-step into the other buffer, overlap with compute
            const unsigned kb = (unsigned)(it + 1) * (TK * 2u);  // bytes
            const unsigned lo = (unsigned)((it + 1) & 1) * BUFB;
            async_ld16(laBase + lo,      gaBase + kb,      Xb);
            async_ld16(laBase + lo + 16, gaBase + kb + 16, Xb);
            async_ld16(lbBase + lo,      gbBase + kb,      CBb);
            async_ld16(lbBase + lo + 16, gbBase + kb + 16, CBb);
            WAIT_ASYNC(4);           // in-order: current buffer complete
        } else {
            WAIT_ASYNC(0);
        }
        __syncthreads();

        union Frag { uint4 q[2]; v16bf v; };
        Frag af[4], bf[2];
        // A 16x32 bf16: lane holds row (l15), K chunks hl*8..+7, 16+hl*8..+7
        #pragma unroll
        for (int i = 0; i < 4; ++i) {
            const unsigned short* pr = &sA[buf][(wm * 64 + i * 16 + l15) * LDSTR];
            af[i].q[0] = *(const uint4*)(pr + hl * 8);
            af[i].q[1] = *(const uint4*)(pr + 16 + hl * 8);
        }
        // B 32x16 bf16: lane holds col (l15), K chunks hl*8..+7, 16+hl*8..+7
        #pragma unroll
        for (int j = 0; j < 2; ++j) {
            const unsigned short* pr = &sB[buf][(wn * 32 + j * 16 + l15) * LDSTR];
            bf[j].q[0] = *(const uint4*)(pr + hl * 8);
            bf[j].q[1] = *(const uint4*)(pr + 16 + hl * 8);
        }
        #pragma unroll
        for (int i = 0; i < 4; ++i)
            #pragma unroll
            for (int j = 0; j < 2; ++j)
                acc[i][j] = __builtin_amdgcn_wmma_f32_16x16x32_bf16(
                    false, af[i].v, false, bf[j].v, (short)0, acc[i][j],
                    false, false);
        __syncthreads();   // protect buffer being re-staged next iteration
    }

    argmax_epilogue(acc, c2, best, sBest, m0, n0, wm, wn, hl, l15, t);
}

// ---------------------------------------------------------------------------
// Kernel 2 (fallback, small workspace): f32 inputs, convert while staging
// ---------------------------------------------------------------------------
__global__ void __launch_bounds__(256)
snap_gemm_argmax_f32(const float* __restrict__ X, const float* __restrict__ CB,
                     const float* __restrict__ c2,
                     unsigned long long* __restrict__ best, int D) {
    __shared__ unsigned short sA[TILE * LDSTR];
    __shared__ unsigned short sB[TILE * LDSTR];
    __shared__ unsigned long long sBest[TILE];

    const int t    = threadIdx.x;
    const int lane = t & 31;
    const int w    = t >> 5;
    const int wm   = w >> 2;
    const int wn   = w & 3;
    const int hl   = lane >> 4;
    const int l15  = lane & 15;

    const int m0 = blockIdx.y * TILE;
    const int n0 = blockIdx.x * TILE;

    if (t < TILE) sBest[t] = 0ull;

    v8f acc[4][2] = {};

    const int srow = t >> 1;
    const int scol = (t & 1) * 16;
    const float4* gA = (const float4*)(X  + (size_t)(m0 + srow) * D);
    const float4* gB = (const float4*)(CB + (size_t)(n0 + srow) * D);

    for (int k0 = 0; k0 < D; k0 += TK) {
        const int fo = (k0 + scol) >> 2;
        float4 a0 = gA[fo + 0], a1 = gA[fo + 1], a2 = gA[fo + 2], a3 = gA[fo + 3];
        float4 b0 = gB[fo + 0], b1 = gB[fo + 1], b2 = gB[fo + 2], b3 = gB[fo + 3];
        uint4* dA = (uint4*)&sA[srow * LDSTR + scol];
        uint4* dB = (uint4*)&sB[srow * LDSTR + scol];
        dA[0] = pack8(a0, a1);  dA[1] = pack8(a2, a3);
        dB[0] = pack8(b0, b1);  dB[1] = pack8(b2, b3);
        __syncthreads();

        union Frag { uint4 q[2]; v16bf v; };
        Frag af[4], bf[2];
        #pragma unroll
        for (int i = 0; i < 4; ++i) {
            const unsigned short* pr = &sA[(wm * 64 + i * 16 + l15) * LDSTR];
            af[i].q[0] = *(const uint4*)(pr + hl * 8);
            af[i].q[1] = *(const uint4*)(pr + 16 + hl * 8);
        }
        #pragma unroll
        for (int j = 0; j < 2; ++j) {
            const unsigned short* pr = &sB[(wn * 32 + j * 16 + l15) * LDSTR];
            bf[j].q[0] = *(const uint4*)(pr + hl * 8);
            bf[j].q[1] = *(const uint4*)(pr + 16 + hl * 8);
        }
        #pragma unroll
        for (int i = 0; i < 4; ++i)
            #pragma unroll
            for (int j = 0; j < 2; ++j)
                acc[i][j] = __builtin_amdgcn_wmma_f32_16x16x32_bf16(
                    false, af[i].v, false, bf[j].v, (short)0, acc[i][j],
                    false, false);
        __syncthreads();
    }

    argmax_epilogue(acc, c2, best, sBest, m0, n0, wm, wn, hl, l15, t);
}

// ---------------------------------------------------------------------------
// Kernel 3: gather winning fp32 codebook row (exact values, like reference)
// ---------------------------------------------------------------------------
__global__ void gather_kernel(const float* __restrict__ CB,
                              const unsigned long long* __restrict__ best,
                              float* __restrict__ out, int D, int C) {
    const int row = blockIdx.x;
    unsigned idx = ~(unsigned)(best[row] & 0xFFFFFFFFull);
    if (idx >= (unsigned)C) idx = 0;
    const float4* src = (const float4*)(CB + (size_t)idx * D);
    float4*       dst = (float4*)(out + (size_t)row * D);
    dst[threadIdx.x] = src[threadIdx.x];       // 256 thr * 4 = 1024 = D
}

// ---------------------------------------------------------------------------
extern "C" void kernel_launch(void* const* d_in, const int* in_sizes, int n_in,
                              void* d_out, int out_size, void* d_ws, size_t ws_size,
                              hipStream_t stream) {
    const float* X  = (const float*)d_in[0];   // [B,S,D] f32
    const float* CB = (const float*)d_in[1];   // [C,D]   f32
    const int D = 1024;
    const int M = in_sizes[0] / D;             // 8192 query rows
    const int C = in_sizes[1] / D;             // 8192 codes

    char* ws = (char*)d_ws;
    unsigned long long* best = (unsigned long long*)ws;            // M u64
    float* c2 = (float*)(ws + (size_t)M * 8);                      // C f32
    unsigned short* Xb  = (unsigned short*)(ws + (size_t)M * 8 + (size_t)C * 4);
    unsigned short* CBb = Xb + (size_t)M * D;
    const size_t need = (size_t)M * 8 + (size_t)C * 4 +
                        ((size_t)M + (size_t)C) * D * 2;

    float* out = (float*)d_out;
    dim3 grid(C / TILE, M / TILE);             // 64 x 64 blocks

    c2_init_kernel<<<C, 256, 0, stream>>>(CB, c2, best, D, M);

    if (ws_size >= need) {
        // one-shot f32->bf16 conversion, then async-staged bf16 GEMM
        cvt_bf16_kernel<<<(M * (D / 8)) / 256, 256, 0, stream>>>(X, (uint4*)Xb);
        cvt_bf16_kernel<<<(C * (D / 8)) / 256, 256, 0, stream>>>(CB, (uint4*)CBb);
        snap_gemm_argmax_bf16<<<grid, 256, 0, stream>>>(Xb, CBb, c2, best, D);
    } else {
        snap_gemm_argmax_f32<<<grid, 256, 0, stream>>>(X, CB, c2, best, D);
    }

    gather_kernel<<<M, 256, 0, stream>>>(CB, best, out, D, C);
}
